// TTTLinear_75754633167130
// MI455X (gfx1250) — compile-verified, matrix operand order checked by reading, not run on hardware
//
#include <hip/hip_runtime.h>
#include <hip/hip_bf16.h>

// Problem constants (reference: B=2, T=2048, C=768, H=12, D=64)
#define BB 2
#define TT 2048
#define CC 768
#define HH 12
#define DD 64
#define INNER_LR 0.01f
#define NCHUNK (TT / 16)

typedef __attribute__((ext_vector_type(2)))  float  v2f;
typedef __attribute__((ext_vector_type(8)))  float  v8f;
typedef __attribute__((ext_vector_type(8)))  __bf16 v8bf;
typedef __attribute__((ext_vector_type(16))) __bf16 v16bf;
typedef __attribute__((ext_vector_type(4)))  unsigned int u32x4;
typedef __attribute__((ext_vector_type(8)))  int i32x8;
typedef __attribute__((ext_vector_type(4)))  int i32x4;

static __device__ __forceinline__ v8f wmma_f32x4(v2f a, v2f b, v8f c) {
  // D = A(16x4 f32) * B(4x16 f32) + C(16x16 f32)
  return __builtin_amdgcn_wmma_f32_16x16x4_f32(false, a, false, b, (short)0, c,
                                               false, false);
}

static __device__ __forceinline__ v8f wmma_bf16(v16bf a, v16bf b, v8f c) {
  // D = A(16x32 bf16) * B(32x16 bf16) + C(16x16 f32)
  return __builtin_amdgcn_wmma_f32_16x16x32_bf16(false, a, false, b, (short)0, c,
                                                 false, false);
}

// ---------------------------------------------------------------------------
// TDM: async-load one 16x64 f32 tile (row stride CC elements) into LDS.
// Pad feature inserts 1 DWORD after every 64 DWORDs -> LDS rows of 65 floats
// (bank-conflict-free), matching the [16][65] shared arrays.
// Descriptor per CDNA5 ISA §8 (group0 128b, group1 256b; groups 2/3 unused).
// Toolchain uses the 6-arg builtin form (extra i32x8 group before cpol).
// ---------------------------------------------------------------------------
static __device__ __forceinline__ void tdm_load_tile16x64(const float* gsrc,
                                                          unsigned int lds_byte) {
  unsigned long long ga = (unsigned long long)(uintptr_t)gsrc;
  u32x4 g0 = {1u,                                   // count=1 (valid descriptor)
              lds_byte,                             // lds_addr
              (unsigned int)ga,                     // global_addr[31:0]
              (unsigned int)((ga >> 32) & 0x1FFFFFFu) | (2u << 30)};  // type=2
  // g1 dword0: data_size=4B(2)<<16 | pad_enable<<20 | pad_interval=5(64dw)<<22
  //            | pad_amount=0(1dw)<<25
  i32x8 g1 = {(2 << 16) | (1 << 20) | (5 << 22),    // flags
              64 << 16,                             // tensor_dim0[15:0]=64 @bit48
              16 << 16,                             // tensor_dim1[15:0]=16 @bit80
              64 << 16,                             // tile_dim0=64 @bit112
              16,                                   // tile_dim1=16 @bit128
              CC,                                   // tensor_dim0_stride lo32
              0, 0};
  i32x4 gz4 = {0, 0, 0, 0};
  i32x8 gz8 = {0, 0, 0, 0, 0, 0, 0, 0};
  __builtin_amdgcn_tensor_load_to_lds(g0, g1, gz4, gz4, gz8, 0);
}

// ---------------------------------------------------------------------------
// f32 -> bf16 conversion (grid-stride)
// ---------------------------------------------------------------------------
__global__ void cvt_bf16_kernel(const float* __restrict__ src,
                                __bf16* __restrict__ dst, int n) {
  int i = blockIdx.x * blockDim.x + threadIdx.x;
  int stride = gridDim.x * blockDim.x;
  for (; i < n; i += stride) dst[i] = (__bf16)src[i];
}

// ---------------------------------------------------------------------------
// C[M,N] = A[M,K](row-major, bf16) * B[N,K]^T (row-major, bf16), f32 out.
// Block = 256 threads = 8 waves; each wave computes a 16(M) x 64(N) strip.
// ---------------------------------------------------------------------------
__global__ __launch_bounds__(256) void gemm_nt_bf16_kernel(
    const __bf16* __restrict__ A, const __bf16* __restrict__ Bw,
    float* __restrict__ Cout, int M, int N, int K) {
  const int lane = threadIdx.x & 31;
  const int wave = threadIdx.x >> 5;
  const int r    = lane & 15;
  const int ksel = lane >> 4;
  const int m0 = blockIdx.x * 128 + wave * 16;
  const int n0 = blockIdx.y * 64;

  v8f acc0 = {}, acc1 = {}, acc2 = {}, acc3 = {};

  const __bf16* arow = A  + (size_t)(m0 + r) * K;
  const __bf16* br0  = Bw + (size_t)(n0 +  0 + r) * K;
  const __bf16* br1  = Bw + (size_t)(n0 + 16 + r) * K;
  const __bf16* br2  = Bw + (size_t)(n0 + 32 + r) * K;
  const __bf16* br3  = Bw + (size_t)(n0 + 48 + r) * K;

  for (int k0 = 0; k0 < K; k0 += 32) {
    const int off = k0 + ksel * 8;
    v8bf alo = *(const v8bf*)(arow + off);
    v8bf ahi = *(const v8bf*)(arow + off + 16);
    v16bf af = __builtin_shufflevector(alo, ahi, 0, 1, 2, 3, 4, 5, 6, 7, 8, 9,
                                       10, 11, 12, 13, 14, 15);
    v8bf b0lo = *(const v8bf*)(br0 + off);
    v8bf b0hi = *(const v8bf*)(br0 + off + 16);
    v8bf b1lo = *(const v8bf*)(br1 + off);
    v8bf b1hi = *(const v8bf*)(br1 + off + 16);
    v8bf b2lo = *(const v8bf*)(br2 + off);
    v8bf b2hi = *(const v8bf*)(br2 + off + 16);
    v8bf b3lo = *(const v8bf*)(br3 + off);
    v8bf b3hi = *(const v8bf*)(br3 + off + 16);
    v16bf bf0 = __builtin_shufflevector(b0lo, b0hi, 0, 1, 2, 3, 4, 5, 6, 7, 8,
                                        9, 10, 11, 12, 13, 14, 15);
    v16bf bf1 = __builtin_shufflevector(b1lo, b1hi, 0, 1, 2, 3, 4, 5, 6, 7, 8,
                                        9, 10, 11, 12, 13, 14, 15);
    v16bf bf2 = __builtin_shufflevector(b2lo, b2hi, 0, 1, 2, 3, 4, 5, 6, 7, 8,
                                        9, 10, 11, 12, 13, 14, 15);
    v16bf bf3 = __builtin_shufflevector(b3lo, b3hi, 0, 1, 2, 3, 4, 5, 6, 7, 8,
                                        9, 10, 11, 12, 13, 14, 15);
    acc0 = wmma_bf16(af, bf0, acc0);
    acc1 = wmma_bf16(af, bf1, acc1);
    acc2 = wmma_bf16(af, bf2, acc2);
    acc3 = wmma_bf16(af, bf3, acc3);
  }

#pragma unroll
  for (int vg = 0; vg < 8; ++vg) {
    int m = m0 + vg + 8 * ksel;
    float* crow = Cout + (size_t)m * N + n0 + r;
    crow[0]  = acc0[vg];
    crow[16] = acc1[vg];
    crow[32] = acc2[vg];
    crow[48] = acc3[vg];
  }
}

// ---------------------------------------------------------------------------
// Chunked TTT scan: one block per (b,h), 256 threads = 8 waves.
// Q/K/V chunk tiles are double-buffered and fetched one chunk ahead by the
// Tensor Data Mover (wave 0 issues, s_wait_tensorcnt + barrier publishes).
// Chunk math (exactly equivalent to per-token GD, fp32 WMMA 16x16x4):
//   raw  = K Wt^T - V ; Akk = K K^T ; Aqk = tril_strict(Q K^T)
//   G_s  = -lr*(raw_s + sum_{r<s} Akk[s][r] G_r)   (16-step solve)
//   Out  = Q Wt^T + Aqk G ;  W += G^T K
// ---------------------------------------------------------------------------
__global__ __launch_bounds__(256) void ttt_scan_kernel(
    const float* __restrict__ qf, const float* __restrict__ kf,
    const float* __restrict__ vf, const float* __restrict__ lr_scale,
    float* __restrict__ hid) {
  const int bh = blockIdx.x;
  const int b = bh / HH, h = bh % HH;
  const float lr = INNER_LR * lr_scale[h];
  const int tid  = threadIdx.x;
  const int lane = tid & 31;
  const int wave = tid >> 5;
  const int r    = lane & 15;
  const int hi2  = lane >> 4;
  const int kb   = hi2 * 2;  // f32 WMMA: lanes 0-15 hold K 0,1; 16-31 hold 2,3

  __shared__ float Wst[DD][DD + 1];
  __shared__ float Qc[2][16][DD + 1];
  __shared__ float Kc[2][16][DD + 1];
  __shared__ float Vc[2][16][DD + 1];
  __shared__ float G[16][DD + 1];
  __shared__ float Akk[16][17];
  __shared__ float Amask[16][17];

  for (int i = tid; i < DD * (DD + 1); i += 256) (&Wst[0][0])[i] = 0.0f;

  const size_t base = (size_t)b * TT * CC + (size_t)h * DD;

  // TDM prologue: fetch chunk 0 into buffer 0 (wave 0 only; per-wave DMA op).
  if (wave == 0) {
    tdm_load_tile16x64(qf + base, (unsigned int)(uintptr_t)&Qc[0][0][0]);
    tdm_load_tile16x64(kf + base, (unsigned int)(uintptr_t)&Kc[0][0][0]);
    tdm_load_tile16x64(vf + base, (unsigned int)(uintptr_t)&Vc[0][0][0]);
  }
  __syncthreads();

  for (int c = 0; c < NCHUNK; ++c) {
    const int t0 = c * 16;
    const int cb = c & 1;
    const int nb = cb ^ 1;

    // Prefetch chunk c+1 into the other buffer, then wait for chunk c's
    // 3 TDM ops (in-order completion; <=3 outstanding for the prefetch).
    if (wave == 0) {
      if (c + 1 < NCHUNK) {
        const size_t nxt = base + (size_t)(t0 + 16) * CC;
        tdm_load_tile16x64(qf + nxt, (unsigned int)(uintptr_t)&Qc[nb][0][0]);
        tdm_load_tile16x64(kf + nxt, (unsigned int)(uintptr_t)&Kc[nb][0][0]);
        tdm_load_tile16x64(vf + nxt, (unsigned int)(uintptr_t)&Vc[nb][0][0]);
        __builtin_amdgcn_s_wait_tensorcnt(3);
      } else {
        __builtin_amdgcn_s_wait_tensorcnt(0);
      }
    }
    __syncthreads();

    v8f accQ = {};  // QW fragment, lives across phases in waves 4-7
    if (wave < 4) {
      // KW tile: rows s, cols i0..i0+15 ; raw = KW - V into G
      const int i0 = wave * 16;
      v8f acc = {};
#pragma unroll
      for (int j0 = 0; j0 < DD; j0 += 4) {
        v2f a  = {Kc[cb][r][j0 + kb], Kc[cb][r][j0 + kb + 1]};
        v2f bb = {Wst[i0 + r][j0 + kb], Wst[i0 + r][j0 + kb + 1]};
        acc = wmma_f32x4(a, bb, acc);
      }
#pragma unroll
      for (int vg = 0; vg < 8; ++vg) {
        int s = vg + 8 * hi2;
        G[s][i0 + r] = acc[vg] - Vc[cb][s][i0 + r];
      }
      if (wave == 1) {  // Akk = K K^T  (A and B fragments coincide)
        v8f ak = {};
#pragma unroll
        for (int j0 = 0; j0 < DD; j0 += 4) {
          v2f a = {Kc[cb][r][j0 + kb], Kc[cb][r][j0 + kb + 1]};
          ak = wmma_f32x4(a, a, ak);
        }
#pragma unroll
        for (int vg = 0; vg < 8; ++vg) Akk[vg + 8 * hi2][r] = ak[vg];
      }
    } else {
      // QW tile (uses pre-update W — "read BEFORE update")
      const int i0 = (wave - 4) * 16;
#pragma unroll
      for (int j0 = 0; j0 < DD; j0 += 4) {
        v2f a  = {Qc[cb][r][j0 + kb], Qc[cb][r][j0 + kb + 1]};
        v2f bb = {Wst[i0 + r][j0 + kb], Wst[i0 + r][j0 + kb + 1]};
        accQ = wmma_f32x4(a, bb, accQ);
      }
      if (wave == 5) {  // Aqk = Q K^T, store strict-lower mask
        v8f aq = {};
#pragma unroll
        for (int j0 = 0; j0 < DD; j0 += 4) {
          v2f a  = {Qc[cb][r][j0 + kb], Qc[cb][r][j0 + kb + 1]};
          v2f bb = {Kc[cb][r][j0 + kb], Kc[cb][r][j0 + kb + 1]};
          aq = wmma_f32x4(a, bb, aq);
        }
#pragma unroll
        for (int vg = 0; vg < 8; ++vg) {
          int s = vg + 8 * hi2;
          Amask[s][r] = (r < s) ? aq[vg] : 0.0f;
        }
      }
    }
    __syncthreads();

    // ---- 16-step triangular solve; thread d owns feature column d ----
    if (tid < DD) {
      const int d = tid;
      G[0][d] = -lr * G[0][d];
      for (int s = 1; s < 16; ++s) {
        float acc = G[s][d];
        for (int rr = 0; rr < s; ++rr) acc += Akk[s][rr] * G[rr][d];
        G[s][d] = -lr * acc;
      }
    }
    __syncthreads();

    if (wave >= 4) {
      // Out = QW + tril_strict(Aqk) * G ; store to hid
      const int i0 = (wave - 4) * 16;
#pragma unroll
      for (int s0 = 0; s0 < 16; s0 += 4) {
        v2f a  = {Amask[r][s0 + kb], Amask[r][s0 + kb + 1]};
        v2f bb = {G[s0 + kb][i0 + r], G[s0 + kb + 1][i0 + r]};
        accQ = wmma_f32x4(a, bb, accQ);
      }
#pragma unroll
      for (int vg = 0; vg < 8; ++vg) {
        int s = vg + 8 * hi2;
        hid[base + (size_t)(t0 + s) * CC + i0 + r] = accQ[vg];
      }
    } else {
      // W += G^T K  (each wave owns one 64x16 column strip: 4 tiles)
#pragma unroll
      for (int tl = 0; tl < 4; ++tl) {
        const int i0 = tl * 16;    // W row tile
        const int j0 = wave * 16;  // W col tile
        v8f cf;
#pragma unroll
        for (int vg = 0; vg < 8; ++vg) cf[vg] = Wst[i0 + vg + 8 * hi2][j0 + r];
#pragma unroll
        for (int s0 = 0; s0 < 16; s0 += 4) {
          v2f a  = {G[s0 + kb][i0 + r], G[s0 + kb + 1][i0 + r]};   // (G^T) frag
          v2f bb = {Kc[cb][s0 + kb][j0 + r], Kc[cb][s0 + kb + 1][j0 + r]};
          cf = wmma_f32x4(a, bb, cf);
        }
#pragma unroll
        for (int vg = 0; vg < 8; ++vg) Wst[i0 + vg + 8 * hi2][j0 + r] = cf[vg];
      }
    }
    __syncthreads();
  }
}

// ---------------------------------------------------------------------------
extern "C" void kernel_launch(void* const* d_in, const int* in_sizes, int n_in,
                              void* d_out, int out_size, void* d_ws,
                              size_t ws_size, hipStream_t stream) {
  const float* x  = (const float*)d_in[0];
  const float* Wq = (const float*)d_in[1];
  const float* Wk = (const float*)d_in[2];
  const float* Wv = (const float*)d_in[3];
  const float* Wo = (const float*)d_in[4];
  const float* lr = (const float*)d_in[5];

  const size_t NX = (size_t)BB * TT * CC;  // 3,145,728 activations
  const size_t NW = (size_t)CC * CC;       // 589,824 weights

  char* p = (char*)d_ws;
  float* qf   = (float*)p;  p += NX * sizeof(float);
  float* kf   = (float*)p;  p += NX * sizeof(float);
  float* vf   = (float*)p;  p += NX * sizeof(float);
  float* hidf = (float*)p;  p += NX * sizeof(float);
  __bf16* xb   = (__bf16*)p; p += NX * sizeof(__bf16);
  __bf16* wqb  = (__bf16*)p; p += NW * sizeof(__bf16);
  __bf16* wkb  = (__bf16*)p; p += NW * sizeof(__bf16);
  __bf16* wvb  = (__bf16*)p; p += NW * sizeof(__bf16);
  __bf16* wob  = (__bf16*)p; p += NW * sizeof(__bf16);
  __bf16* hidb = (__bf16*)p; p += NX * sizeof(__bf16);

  // 1) convert inputs to bf16 for the projection GEMMs
  cvt_bf16_kernel<<<2048, 256, 0, stream>>>(x, xb, (int)NX);
  cvt_bf16_kernel<<<1024, 256, 0, stream>>>(Wq, wqb, (int)NW);
  cvt_bf16_kernel<<<1024, 256, 0, stream>>>(Wk, wkb, (int)NW);
  cvt_bf16_kernel<<<1024, 256, 0, stream>>>(Wv, wvb, (int)NW);
  cvt_bf16_kernel<<<1024, 256, 0, stream>>>(Wo, wob, (int)NW);

  // 2) q/k/v = x @ W^T   (M=B*T=4096, N=C=768, K=C=768)
  dim3 ggrid(4096 / 128, 768 / 64);
  gemm_nt_bf16_kernel<<<ggrid, 256, 0, stream>>>(xb, wqb, qf, 4096, 768, 768);
  gemm_nt_bf16_kernel<<<ggrid, 256, 0, stream>>>(xb, wkb, kf, 4096, 768, 768);
  gemm_nt_bf16_kernel<<<ggrid, 256, 0, stream>>>(xb, wvb, vf, 4096, 768, 768);

  // 3) chunked TTT scan (one block per (b,h)), TDM double-buffered
  ttt_scan_kernel<<<BB * HH, 256, 0, stream>>>(qf, kf, vf, lr, hidf);

  // 4) out = hid @ Wo^T
  cvt_bf16_kernel<<<2048, 256, 0, stream>>>(hidf, hidb, (int)NX);
  gemm_nt_bf16_kernel<<<ggrid, 256, 0, stream>>>(hidb, wob, (float*)d_out, 4096,
                                                 768, 768);
}
